// UNetInnerProd_88819923682047
// MI455X (gfx1250) — compile-verified, matrix operand order checked by reading, not run on hardware
//
#include <hip/hip_runtime.h>

// ---------------- problem constants ----------------
#define W_IMG   96
#define H_IMG   72
#define N_PIX   (W_IMG * H_IMG)     // 6912
#define C_FEAT  64
#define B_BATCH 2
#define NT      (N_PIX / 16)        // 432 tiles of 16 pixels
#define FRAG_HALVES 1024            // one 16x64 f16 fragment (A or B) per tile
#define DIST_COEF 0.1f
#define LOG2E 1.44269504088896340736f

typedef _Float16 f16;
typedef __attribute__((ext_vector_type(16))) _Float16 v16h;
typedef __attribute__((ext_vector_type(8)))  float    v8f;
typedef __attribute__((ext_vector_type(4)))  int      v4i;

// builtin parameter types: int4* in global (AS1) and LDS (AS3) address spaces
typedef __attribute__((address_space(1))) v4i as1_v4i;
typedef __attribute__((address_space(3))) v4i as3_v4i;

// ---------------- CDNA5 async global->LDS helpers ----------------
__device__ __forceinline__ void async_copy_b128(const void* g, void* l) {
#if defined(__has_builtin) && __has_builtin(__builtin_amdgcn_global_load_async_to_lds_b128)
    __builtin_amdgcn_global_load_async_to_lds_b128((as1_v4i*)g, (as3_v4i*)l, 0, 0);
#else
    unsigned loff = (unsigned)(unsigned long long)(__attribute__((address_space(3))) void*)l;
    unsigned long long ga = (unsigned long long)g;
    asm volatile("global_load_async_to_lds_b128 %0, %1, off" :: "v"(loff), "v"(ga) : "memory");
#endif
}

template <int N>
__device__ __forceinline__ void wait_asynccnt() {
#if defined(__has_builtin) && __has_builtin(__builtin_amdgcn_s_wait_asynccnt)
    __builtin_amdgcn_s_wait_asynccnt(N);
#else
    asm volatile("s_wait_asynccnt %0" :: "n"(N) : "memory");
#endif
}

// CDNA5 WMMA 16-bit A/B fragment addressing (ISA 7.12.2).
// 16x32 f16 fragment: lanes 0-15 hold K'=0..7 (halves 0-7) and K'=16..23 (halves 8-15)
// for row/col M; lanes 16-31 hold K'=8..15 and K'=24..31.  Two K-halves (K=0..31,
// K=32..63) stored back to back (512 halves each).  Lane data contiguous (v16h/lane).
__device__ __forceinline__ int frag_idx(int c /*K 0..63*/, int m /*row or col 0..15*/) {
    int kf   = c >> 5;          // which 16x32 sub-fragment
    int kp   = c & 31;
    int lane = m + (((kp >> 3) & 1) << 4);
    int h    = (kp & 7) + ((kp >> 4) << 3);
    return kf * 512 + lane * 16 + h;
}

// ---------------- kernel 0: zero accumulators ----------------
__global__ void unet_init_kernel(float* __restrict__ acc) {
    if (threadIdx.x < B_BATCH) acc[threadIdx.x] = 0.0f;
}

// ---------------- kernel 1: back-project + normalize + pack fragments ----------------
__global__ void __launch_bounds__(256)
unet_prep_kernel(const float* __restrict__ f1, const float* __restrict__ f2,
                 const float* __restrict__ d1, const float* __restrict__ d2,
                 const float* __restrict__ pose,
                 float* __restrict__ xyz1, float* __restrict__ xyz2t,
                 f16* __restrict__ aF, f16* __restrict__ bF) {
    int g = blockIdx.x * blockDim.x + threadIdx.x;
    if (g >= B_BATCH * N_PIX) return;
    int b = g / N_PIX;
    int p = g - b * N_PIX;
    int u = p % W_IMG;
    int v = p / W_IMG;

    // invK * [u,v,1] for CARLA K=[[48,48,0],[36,0,48],[1,0,0]] -> (1,(u-48)/48,(v-36)/48)
    float m0 = 1.0f;
    float m1 = ((float)u - 48.0f) * (1.0f / 48.0f);
    float m2 = ((float)v - 36.0f) * (1.0f / 48.0f);

    float dd1 = d1[b * N_PIX + p];
    float dd2 = d2[b * N_PIX + p];
    float ax = m0 * dd1, ay = m1 * dd1, az = m2 * dd1;
    float cx = m0 * dd2, cy = m1 * dd2, cz = m2 * dd2;

    const float* P = pose + b * 16;
    float tx = P[0] * cx + P[1] * cy + P[2]  * cz + P[3];
    float ty = P[4] * cx + P[5] * cy + P[6]  * cz + P[7];
    float tz = P[8] * cx + P[9] * cy + P[10] * cz + P[11];

    xyz1[(b * 3 + 0) * N_PIX + p] = ax;
    xyz1[(b * 3 + 1) * N_PIX + p] = ay;
    xyz1[(b * 3 + 2) * N_PIX + p] = az;
    xyz2t[(b * 3 + 0) * N_PIX + p] = tx;
    xyz2t[(b * 3 + 1) * N_PIX + p] = ty;
    xyz2t[(b * 3 + 2) * N_PIX + p] = tz;

    // feature column norms (sum over C, stride N -> coalesced across threads)
    const float* F1 = f1 + (size_t)b * C_FEAT * N_PIX + p;
    const float* F2 = f2 + (size_t)b * C_FEAT * N_PIX + p;
    float s1 = 0.0f, s2 = 0.0f;
    #pragma unroll 8
    for (int c = 0; c < C_FEAT; ++c) {
        float x = F1[c * N_PIX]; s1 = fmaf(x, x, s1);
        float y = F2[c * N_PIX]; s2 = fmaf(y, y, s2);
    }
    float r1 = rsqrtf(fmaxf(s1, 1e-24f));
    float r2 = rsqrtf(fmaxf(s2, 1e-24f));

    int t = p >> 4;      // tile
    int m = p & 15;      // row (A) / col (B) within tile
    f16* A  = aF + (size_t)(b * NT + t) * FRAG_HALVES;
    f16* Bf = bF + (size_t)(b * NT + t) * FRAG_HALVES;
    #pragma unroll 8
    for (int c = 0; c < C_FEAT; ++c) {
        int idx = frag_idx(c, m);
        A[idx]  = (f16)(F1[c * N_PIX] * r1);
        Bf[idx] = (f16)(F2[c * N_PIX] * r2);
    }
}

// ---------------- kernel 2: WMMA gram tiles * distance kernel, reduce ----------------
__global__ void __launch_bounds__(256)
unet_gemm_kernel(const f16* __restrict__ aF, const f16* __restrict__ bF,
                 const float* __restrict__ xyz1, const float* __restrict__ xyz2t,
                 float* __restrict__ acc) {
    __shared__ uint4 smemB[2][1024];     // double-buffered 8 B-fragments (2 x 16 KB)
    __shared__ float smemP[3][256];      // xyz2t for this block's 256 columns (3 KB)

    const int lane = threadIdx.x & 31;
    const int wave = threadIdx.x >> 5;
    const int b    = blockIdx.z;
    const int ti   = blockIdx.x * 8 + wave;        // this wave's row tile
    const int tj0  = blockIdx.y * 16;              // 16 column tiles per block

    // -- kick off async global->LDS staging of BOTH B-fragment chunks up front --
    // 16 KB per chunk, 256 threads x 4 x b128 each; ASYNCcnt-tracked, no VGPR traffic.
    #pragma unroll
    for (int chunk = 0; chunk < 2; ++chunk) {
        const char* src = (const char*)(bF + (size_t)(b * NT + tj0 + chunk * 8) * FRAG_HALVES);
        char* dst = (char*)smemB[chunk];
        #pragma unroll
        for (int k = 0; k < 4; ++k) {
            int e = threadIdx.x + k * 256;
            async_copy_b128(src + e * 16, dst + e * 16);
        }
    }

    // -- stage this block's 256 xyz2t columns into LDS (overlaps with async DMA) --
    {
        int col = threadIdx.x;                    // 0..255
        int j   = tj0 * 16 + col;
        smemP[0][col] = xyz2t[(size_t)(b * 3 + 0) * N_PIX + j];
        smemP[1][col] = xyz2t[(size_t)(b * 3 + 1) * N_PIX + j];
        smemP[2][col] = xyz2t[(size_t)(b * 3 + 2) * N_PIX + j];
    }

    // -- A fragments for row tile ti: loaded once, reused across 16 column tiles --
    const char* aBase = (const char*)(aF + (size_t)(b * NT + ti) * FRAG_HALVES);
    v16h a0 = *(const v16h*)(aBase + lane * 32);
    v16h a1 = *(const v16h*)(aBase + 1024 + lane * 32);

    // -- xyz1 rows this lane's accumulator elements touch: i = ibase + r, r=0..7 --
    const int ibase = ti * 16 + ((lane >> 4) << 3);
    float x1r[8], y1r[8], z1r[8];
    {
        const float4* px = (const float4*)(xyz1 + (size_t)(b * 3 + 0) * N_PIX + ibase);
        const float4* py = (const float4*)(xyz1 + (size_t)(b * 3 + 1) * N_PIX + ibase);
        const float4* pz = (const float4*)(xyz1 + (size_t)(b * 3 + 2) * N_PIX + ibase);
        float4 xa = px[0], xb = px[1];
        float4 ya = py[0], yb = py[1];
        float4 za = pz[0], zb = pz[1];
        x1r[0]=xa.x; x1r[1]=xa.y; x1r[2]=xa.z; x1r[3]=xa.w; x1r[4]=xb.x; x1r[5]=xb.y; x1r[6]=xb.z; x1r[7]=xb.w;
        y1r[0]=ya.x; y1r[1]=ya.y; y1r[2]=ya.z; y1r[3]=ya.w; y1r[4]=yb.x; y1r[5]=yb.y; y1r[6]=yb.z; y1r[7]=yb.w;
        z1r[0]=za.x; z1r[1]=za.y; z1r[2]=za.z; z1r[3]=za.w; z1r[4]=zb.x; z1r[5]=zb.y; z1r[6]=zb.z; z1r[7]=zb.w;
    }

    float lsum = 0.0f;

    #pragma unroll
    for (int chunk = 0; chunk < 2; ++chunk) {
        // async loads complete in order: <=4 outstanding releases chunk 0 while
        // chunk 1's DMA still overlaps compute; <=0 releases chunk 1.
        if (chunk == 0) wait_asynccnt<4>(); else wait_asynccnt<0>();
        __syncthreads();

        #pragma unroll 2
        for (int jj = 0; jj < 8; ++jj) {
            const char* bBase = (const char*)smemB[chunk] + jj * 2048;
            v16h b0 = *(const v16h*)(bBase + lane * 32);
            v16h b1 = *(const v16h*)(bBase + 1024 + lane * 32);

            v8f c = {};
            c = __builtin_amdgcn_wmma_f32_16x16x32_f16(false, a0, false, b0, (short)0, c, false, false);
            c = __builtin_amdgcn_wmma_f32_16x16x32_f16(false, a1, false, b1, (short)0, c, false, false);

            // this lane's column within the block: all from LDS, no global traffic
            const int col = (chunk * 8 + jj) * 16 + (lane & 15);
            float p2x = smemP[0][col];
            float p2y = smemP[1][col];
            float p2z = smemP[2][col];

            #pragma unroll
            for (int r = 0; r < 8; ++r) {
                float dx = x1r[r] - p2x;
                float dy = y1r[r] - p2y;
                float dz = z1r[r] - p2z;
                float d2v = fmaf(dx, dx, fmaf(dy, dy, dz * dz));
                float d   = __builtin_amdgcn_sqrtf(fmaxf(d2v, 1e-12f));
                float kv  = __builtin_amdgcn_exp2f(d * (-DIST_COEF * LOG2E));
                lsum = fmaf(kv, c[r], lsum);
            }
        }
    }

    // wave32 reduction, then one atomic per wave
    #pragma unroll
    for (int off = 16; off > 0; off >>= 1) lsum += __shfl_xor(lsum, off, 32);
    if (lane == 0) atomicAdd(&acc[b], lsum);
}

// ---------------- kernel 3: finalize ----------------
__global__ void unet_final_kernel(const float* __restrict__ acc, float* __restrict__ out) {
    float s = acc[0] + acc[1];
    float v = -s * (1.0f / ((float)N_PIX * (float)N_PIX));
    out[0] = v;      // final_loss
    out[1] = v;      // inner_neg
    out[2] = 0.0f;   // fea_norm_sum
}

// ---------------- launcher ----------------
extern "C" void kernel_launch(void* const* d_in, const int* in_sizes, int n_in,
                              void* d_out, int out_size, void* d_ws, size_t ws_size,
                              hipStream_t stream) {
    const float* feature1 = (const float*)d_in[0];
    const float* feature2 = (const float*)d_in[1];
    const float* depth1   = (const float*)d_in[2];
    const float* depth2   = (const float*)d_in[3];
    const float* pose1_2  = (const float*)d_in[4];
    // d_in[5], d_in[6] = img1, img2 -> unused (color_in_cost=False)
    float* out = (float*)d_out;

    // workspace carve-up (all offsets 256B aligned; total ~3.7 MB)
    char* ws = (char*)d_ws;
    const size_t xyzBytes  = sizeof(float) * B_BATCH * 3 * N_PIX;              // 165888
    const size_t fragBytes = sizeof(f16) * (size_t)B_BATCH * NT * FRAG_HALVES; // 1769472
    float* acc   = (float*)ws;                                   // 2 floats
    float* xyz1  = (float*)(ws + 256);
    float* xyz2t = (float*)(ws + 256 + xyzBytes);
    f16*   aF    = (f16*)(ws + 256 + 2 * xyzBytes);
    f16*   bF    = (f16*)(ws + 256 + 2 * xyzBytes + fragBytes);

    unet_init_kernel<<<1, 32, 0, stream>>>(acc);

    int prepThreads = B_BATCH * N_PIX;                           // 13824
    unet_prep_kernel<<<(prepThreads + 255) / 256, 256, 0, stream>>>(
        feature1, feature2, depth1, depth2, pose1_2, xyz1, xyz2t, aF, bF);

    // 432 row tiles / 8 per block = 54; 432 col tiles / 16 per block = 27; batch = 2
    dim3 grid(NT / 8, NT / 16, B_BATCH);
    unet_gemm_kernel<<<grid, 256, 0, stream>>>(aF, bF, xyz1, xyz2t, acc);

    unet_final_kernel<<<1, 1, 0, stream>>>(acc, out);
}
